// CNNLSTMSeq2SeqModel_75359496176080
// MI455X (gfx1250) — compile-verified, matrix operand order checked by reading, not run on hardware
//
#include <hip/hip_runtime.h>
#include <hip/hip_bf16.h>

// ---------------- types for WMMA ----------------
typedef __bf16 v16bf __attribute__((ext_vector_type(16)));
typedef float  v8f   __attribute__((ext_vector_type(8)));

union FragBF {
    v16bf v;
    uint4 u[2];   // 2 x 16 bytes = 16 bf16
};

// ---------------- problem constants ----------------
constexpr int BATCH = 512;
constexpr int SLEN  = 1024;
constexpr int CIN   = 16;
constexpr int COUT  = 64;    // conv out channels == encoder LSTM input size
constexpr int TPOOL = 512;   // pooled sequence length
constexpr int HID   = 512;
constexpr int G4    = 2048;  // 4*HID gate width
constexpr int OUTS  = 64;

__device__ __forceinline__ float sigm(float v) {
    return 1.0f / (1.0f + __expf(-v));
}

// ---------------- conv + relu + maxpool -> bf16 features [T][B][COUT] ----------------
__global__ void conv_relu_pool(const float* __restrict__ x,
                               const float* __restrict__ w,
                               const float* __restrict__ bias,
                               __hip_bfloat16* __restrict__ seq) {
    __shared__ float wl[COUT * CIN * 5];
    __shared__ float bl[COUT];
    for (int i = threadIdx.x; i < COUT * CIN * 5; i += blockDim.x) wl[i] = w[i];
    if (threadIdx.x < COUT) bl[threadIdx.x] = bias[threadIdx.x];
    __syncthreads();

    const int c    = threadIdx.x & 63;
    const int pair = (blockIdx.x << 2) | (threadIdx.x >> 6); // t*BATCH + b
    const int t    = pair >> 9;
    const int b    = pair & 511;

    float m = 0.0f;  // relu outputs are >= 0, so 0 is a safe identity for the max
    #pragma unroll
    for (int soff = 0; soff < 2; ++soff) {
        const int s = 2 * t + soff;
        float acc = bl[c];
        #pragma unroll
        for (int k = 0; k < 5; ++k) {
            const int sr = s + k - 2;
            if (sr >= 0 && sr < SLEN) {
                const float* xr = x + ((size_t)b * SLEN + sr) * CIN;
                const float* wr = wl + (c * CIN) * 5 + k;
                #pragma unroll
                for (int ci = 0; ci < CIN; ++ci)
                    acc += xr[ci] * wr[ci * 5];
            }
        }
        m = fmaxf(m, fmaxf(acc, 0.0f));
    }
    seq[((size_t)t * BATCH + b) * COUT + c] = __float2bfloat16(m);
}

// ---------------- fp32 -> bf16 weight conversion ----------------
__global__ void f32_to_bf16(const float* __restrict__ in,
                            __hip_bfloat16* __restrict__ out, int n) {
    int i = blockIdx.x * blockDim.x + threadIdx.x;
    if (i < n) out[i] = __float2bfloat16(in[i]);
}

// ---------------- state init ----------------
__global__ void init_state(__hip_bfloat16* __restrict__ h, float* __restrict__ c,
                           float* __restrict__ pred, const float* __restrict__ dec_start) {
    int i = blockIdx.x * blockDim.x + threadIdx.x;
    if (i < BATCH * HID) { h[i] = __float2bfloat16(0.0f); c[i] = 0.0f; }
    if (i < BATCH)       pred[i] = dec_start[i];
}

// ---------------- WMMA bf16 GEMM: gates[512 x 2048] = A1*W1^T + A2*W2^T ----------------
// A row-major [M, K] bf16 (K contiguous); W row-major [N, K] bf16 (torch layout).
// grid = (4, 16), block = 256 (8 waves); wave tile 32(M) x 64(N); block tile 128x128.
__global__ void __launch_bounds__(256)
gemm_gates(const __hip_bfloat16* __restrict__ A1,
           const __hip_bfloat16* __restrict__ W1, int K1,
           const __hip_bfloat16* __restrict__ A2,
           const __hip_bfloat16* __restrict__ W2, int K2,
           float* __restrict__ gates) {
    const int lane = threadIdx.x & 31;
    const int wid  = threadIdx.x >> 5;
    const int wm   = wid & 3;                       // 4 wave rows
    const int wn   = wid >> 2;                      // 2 wave cols
    const int M0   = blockIdx.x * 128 + wm * 32;
    const int N0   = blockIdx.y * 128 + wn * 64;
    const int r    = lane & 15;                     // A row / B col within 16-tile
    const int kb   = (lane >> 4) << 3;              // K sub-offset: 0 or 8

    const v8f zero = {0.f, 0.f, 0.f, 0.f, 0.f, 0.f, 0.f, 0.f};
    v8f acc[2][4];
    #pragma unroll
    for (int mi = 0; mi < 2; ++mi)
        #pragma unroll
        for (int ni = 0; ni < 4; ++ni) acc[mi][ni] = zero;

    auto kloop = [&](const __hip_bfloat16* __restrict__ A,
                     const __hip_bfloat16* __restrict__ W, int K) {
        for (int k0 = 0; k0 < K; k0 += 32) {
            FragBF a[2], bw[4];
            #pragma unroll
            for (int mi = 0; mi < 2; ++mi) {
                const __hip_bfloat16* ap = A + (size_t)(M0 + 16 * mi + r) * K + k0 + kb;
                a[mi].u[0] = *(const uint4*)(ap);
                a[mi].u[1] = *(const uint4*)(ap + 16);
            }
            #pragma unroll
            for (int ni = 0; ni < 4; ++ni) {
                const __hip_bfloat16* wp = W + (size_t)(N0 + 16 * ni + r) * K + k0 + kb;
                bw[ni].u[0] = *(const uint4*)(wp);
                bw[ni].u[1] = *(const uint4*)(wp + 16);
            }
            if (k0 + 32 < K) {  // hint next K-chunk into cache (global_prefetch_b8)
                __builtin_prefetch(W + (size_t)(N0 + r) * K + k0 + 32 + kb, 0, 1);
                __builtin_prefetch(A + (size_t)(M0 + r) * K + k0 + 32 + kb, 0, 1);
            }
            #pragma unroll
            for (int mi = 0; mi < 2; ++mi)
                #pragma unroll
                for (int ni = 0; ni < 4; ++ni)
                    acc[mi][ni] = __builtin_amdgcn_wmma_f32_16x16x32_bf16(
                        false, a[mi].v, false, bw[ni].v,
                        (short)0, acc[mi][ni], false, false);
        }
    };

    if (K1 > 0) kloop(A1, W1, K1);   // x-contribution (encoder: K=64)
    kloop(A2, W2, K2);               // h-contribution (K=512)

    const int row_hi = (lane >> 4) << 3;  // 0 or 8
    #pragma unroll
    for (int mi = 0; mi < 2; ++mi)
        #pragma unroll
        for (int ni = 0; ni < 4; ++ni) {
            const int col = N0 + 16 * ni + r;
            #pragma unroll
            for (int v = 0; v < 8; ++v) {
                const int row = M0 + 16 * mi + row_hi + v;
                gates[(size_t)row * G4 + col] = acc[mi][ni][v];
            }
        }
}

// ---------------- LSTM cell (encoder): gates already contain x,W_ih and h,W_hh ----------------
__global__ void lstm_cell_enc(const float* __restrict__ gates,
                              const float* __restrict__ bias,
                              float* __restrict__ c,
                              __hip_bfloat16* __restrict__ h) {
    int idx = blockIdx.x * blockDim.x + threadIdx.x;
    if (idx >= BATCH * HID) return;
    const int b = idx >> 9;
    const int j = idx & 511;
    const float* g = gates + (size_t)b * G4;
    const float gi = g[j]          + bias[j];
    const float gf = g[512 + j]    + bias[512 + j];
    const float gg = g[1024 + j]   + bias[1024 + j];
    const float go = g[1536 + j]   + bias[1536 + j];
    const float cn = sigm(gf) * c[idx] + sigm(gi) * tanhf(gg);
    c[idx] = cn;
    h[idx] = __float2bfloat16(sigm(go) * tanhf(cn));
}

// ---------------- LSTM cell (decoder): adds rank-1 pred * W_ih term ----------------
__global__ void lstm_cell_dec(const float* __restrict__ gates,
                              const float* __restrict__ bias,
                              const float* __restrict__ wih,   // [2048]
                              const float* __restrict__ pred,  // [B]
                              float* __restrict__ c,
                              __hip_bfloat16* __restrict__ h) {
    int idx = blockIdx.x * blockDim.x + threadIdx.x;
    if (idx >= BATCH * HID) return;
    const int b = idx >> 9;
    const int j = idx & 511;
    const float p = pred[b];
    const float* g = gates + (size_t)b * G4;
    const float gi = g[j]          + bias[j]          + p * wih[j];
    const float gf = g[512 + j]    + bias[512 + j]    + p * wih[512 + j];
    const float gg = g[1024 + j]   + bias[1024 + j]   + p * wih[1024 + j];
    const float go = g[1536 + j]   + bias[1536 + j]   + p * wih[1536 + j];
    const float cn = sigm(gf) * c[idx] + sigm(gi) * tanhf(gg);
    c[idx] = cn;
    h[idx] = __float2bfloat16(sigm(go) * tanhf(cn));
}

// ---------------- linear head: pred = h @ head_w^T + head_b; write out column t ----------------
__global__ void head_step(const __hip_bfloat16* __restrict__ h,
                          const float* __restrict__ hw,
                          const float* __restrict__ hb,
                          float* __restrict__ pred,
                          float* __restrict__ out, int t) {
    int b = blockIdx.x * blockDim.x + threadIdx.x;
    if (b >= BATCH) return;
    float s = hb[0];
    const __hip_bfloat16* hr = h + (size_t)b * HID;
    for (int j = 0; j < HID; ++j) s += __bfloat162float(hr[j]) * hw[j];
    pred[b] = s;
    out[(size_t)b * OUTS + t] = s;
}

// ---------------- driver ----------------
extern "C" void kernel_launch(void* const* d_in, const int* in_sizes, int n_in,
                              void* d_out, int out_size, void* d_ws, size_t ws_size,
                              hipStream_t stream) {
    const float* x         = (const float*)d_in[0];
    const float* dec_start = (const float*)d_in[1];
    const float* conv_w    = (const float*)d_in[2];
    const float* conv_b    = (const float*)d_in[3];
    const float* enc_Wih   = (const float*)d_in[4];
    const float* enc_Whh   = (const float*)d_in[5];
    const float* enc_b     = (const float*)d_in[6];
    const float* dec_Wih   = (const float*)d_in[7];
    const float* dec_Whh   = (const float*)d_in[8];
    const float* dec_b     = (const float*)d_in[9];
    const float* head_w    = (const float*)d_in[10];
    const float* head_b    = (const float*)d_in[11];
    float* out = (float*)d_out;

    char* ws = (char*)d_ws;
    size_t off = 0;
    auto alloc = [&](size_t bytes) -> void* {
        void* p = ws + off;
        off += (bytes + 255) & ~(size_t)255;
        return p;
    };

    __hip_bfloat16* seq   = (__hip_bfloat16*)alloc((size_t)TPOOL * BATCH * COUT * 2);
    __hip_bfloat16* wih_e = (__hip_bfloat16*)alloc((size_t)G4 * COUT * 2);
    __hip_bfloat16* whh_e = (__hip_bfloat16*)alloc((size_t)G4 * HID * 2);
    __hip_bfloat16* whh_d = (__hip_bfloat16*)alloc((size_t)G4 * HID * 2);
    float*          gates = (float*)alloc((size_t)BATCH * G4 * 4);
    __hip_bfloat16* hbuf  = (__hip_bfloat16*)alloc((size_t)BATCH * HID * 2);
    float*          cbuf  = (float*)alloc((size_t)BATCH * HID * 4);
    float*          pred  = (float*)alloc((size_t)BATCH * 4);

    // feature extraction + weight precision staging (parallel, one-shot)
    conv_relu_pool<<<TPOOL * BATCH / 4, 256, 0, stream>>>(x, conv_w, conv_b, seq);
    f32_to_bf16<<<(G4 * COUT + 255) / 256, 256, 0, stream>>>(enc_Wih, wih_e, G4 * COUT);
    f32_to_bf16<<<(G4 * HID  + 255) / 256, 256, 0, stream>>>(enc_Whh, whh_e, G4 * HID);
    f32_to_bf16<<<(G4 * HID  + 255) / 256, 256, 0, stream>>>(dec_Whh, whh_d, G4 * HID);
    init_state<<<(BATCH * HID + 255) / 256, 256, 0, stream>>>(hbuf, cbuf, pred, dec_start);

    const dim3 ggrid(4, 16);  // 128x128 block tiles over [512 x 2048]

    // encoder recurrence: 512 sequential steps, weights stay resident in L2
    for (int t = 0; t < TPOOL; ++t) {
        gemm_gates<<<ggrid, 256, 0, stream>>>(seq + (size_t)t * BATCH * COUT, wih_e, COUT,
                                              hbuf, whh_e, HID, gates);
        lstm_cell_enc<<<BATCH * HID / 256, 256, 0, stream>>>(gates, enc_b, cbuf, hbuf);
    }

    // autoregressive decoder
    for (int t = 0; t < OUTS; ++t) {
        gemm_gates<<<ggrid, 256, 0, stream>>>(nullptr, nullptr, 0,
                                              hbuf, whh_d, HID, gates);
        lstm_cell_dec<<<BATCH * HID / 256, 256, 0, stream>>>(gates, dec_b, dec_Wih, pred,
                                                             cbuf, hbuf);
        head_step<<<(BATCH + 255) / 256, 256, 0, stream>>>(hbuf, head_w, head_b, pred, out, t);
    }
}
